// Generator_49117245997364
// MI455X (gfx1250) — matrix operand order repaired and checked by reading, not compiled, not run on hardware
//
#include <hip/hip_runtime.h>

// Problem constants (match reference).
#define BATCH 8192
#define TSTEPS 256
#define DIN 5
#define HH1 10
#define HH2 7
#define HH3 4
#define G1 (4 * HH1) // 40
#define G2 (4 * HH2) // 28
#define G3 (4 * HH3) // 16

typedef __attribute__((ext_vector_type(16))) _Float16 v16h;
typedef __attribute__((ext_vector_type(8)))  _Float16 v8h;
typedef __attribute__((ext_vector_type(8)))  float    v8f;
typedef __attribute__((ext_vector_type(4)))  float    v4f;

#define LOG2E 1.4426950408889634f

// Fast sigmoid: v_mul + v_exp_f32 + v_add + v_rcp_f32 (2 TRANS + 2 VALU).
// Avoids the IEEE division expansion (v_div_scale/v_div_fmas Newton-Raphson)
// that dominated VALU in the previous build. ~1ulp rcp error is far below the
// f16 gate quantization noise.
__device__ __forceinline__ float fsig(float x) {
    float e = __builtin_amdgcn_exp2f(-x * LOG2E);
    return __builtin_amdgcn_rcpf(1.0f + e);
}
__device__ __forceinline__ float ftanh_fast(float x) {
    // tanh(x) = 2*sigmoid(2x) - 1 (stays on the TRANS pipe, co-executes w/ WMMA)
    return __builtin_fmaf(2.0f, fsig(2.0f * x), -1.0f);
}

__device__ __forceinline__ v8f bcast8(float v) {
    v8f r;
#pragma unroll
    for (int i = 0; i < 8; ++i) r[i] = v;
    return r;
}

// Build a persistent f16 B-fragment (32x16 K-major tile of [W;U] stacked, zero padded).
// Layout: column n = lane&15; lanes 0-15 hold K=0..7 (elems 0..7), K=16..23 (elems 8..15),
// lanes 16-31 hold K=8..15 (elems 0..7), K=24..31 (elems 8..15).
__device__ __forceinline__ v16h make_bfrag(const float* __restrict__ W,
                                           const float* __restrict__ U,
                                           int din, int h, int G, int ntile,
                                           int nl, int hl) {
    const int n = ntile * 16 + nl;
    v16h r;
#pragma unroll
    for (int idx = 0; idx < 16; ++idx) {
        const int k = (idx < 8) ? (hl * 8 + idx) : (16 + hl * 8 + (idx - 8));
        float v = 0.0f;
        if (n < G) {
            if (k < din)            v = W[k * G + n];
            else if (k < din + h)   v = U[(k - din) * G + n];
        }
        r[idx] = (_Float16)v;
    }
    return r;
}

// Load an A fragment (16x32 f16, rows = batch) from a zero-padded LDS staging
// buffer (row stride 32 halfs). Hardware A K-map:
//   lanes 0-15 : elems 0..7 = K0..7,  elems 8..15 = K16..23
//   lanes 16-31: elems 0..7 = K8..15, elems 8..15 = K24..31
__device__ __forceinline__ v16h load_afrag(const _Float16* __restrict__ abuf,
                                           int nl, int hl) {
    const _Float16* row = abuf + nl * 32 + hl * 8;
    v8h lo = *(const v8h*)(row);        // 16B aligned -> ds_load_b128
    v8h hi = *(const v8h*)(row + 16);   // 16B aligned -> ds_load_b128
    v16h r;
#pragma unroll
    for (int i = 0; i < 8; ++i) { r[i] = lo[i]; r[i + 8] = hi[i]; }
    return r;
}

// Store a 16x16 f32 C/D tile to LDS in [n][m] order (two ds_store_b128/lane).
// C layout: element r of v8f = (M = r + 8*hl, N = lane&15).
__device__ __forceinline__ void store_z(float* __restrict__ zb, v8f acc,
                                        int ntile, int nl, int hl) {
    float* p = zb + ((ntile * 16 + nl) * 16 + hl * 8);
    v4f lo = {acc[0], acc[1], acc[2], acc[3]};
    v4f hi = {acc[4], acc[5], acc[6], acc[7]};
    *(v4f*)(p)     = lo;
    *(v4f*)(p + 4) = hi;
}

__global__ __launch_bounds__(32)
void lstm3_wmma_kernel(const float* __restrict__ x,
                       const float* __restrict__ W1, const float* __restrict__ U1, const float* __restrict__ b1,
                       const float* __restrict__ W2, const float* __restrict__ U2, const float* __restrict__ b2,
                       const float* __restrict__ W3, const float* __restrict__ U3, const float* __restrict__ b3,
                       const float* __restrict__ Wd, const float* __restrict__ bd,
                       float* __restrict__ out) {
    // A-staging buffers (f16, zero padded K to 32): a1 = [x|h1], a2 = [h1|h2], a3 = [h2|h3]
    __shared__ __align__(16) _Float16 a1[16 * 32];
    __shared__ __align__(16) _Float16 a2[16 * 32];
    __shared__ __align__(16) _Float16 a3[16 * 32];
    // WMMA result scratch, [n][m] f32, up to 3 tiles of 16 columns.
    __shared__ __align__(16) float zbuf[48 * 16];

    const int lane  = threadIdx.x;
    const int nl    = lane & 15;   // batch row within tile / N column
    const int hl    = lane >> 4;   // which half of the wave
    const int bbase = blockIdx.x * 16;

    // Zero staging buffers once: K padding must stay zero forever.
    for (int i = lane; i < 16 * 32; i += 32) {
        a1[i] = (_Float16)0.0f;
        a2[i] = (_Float16)0.0f;
        a3[i] = (_Float16)0.0f;
    }
    __syncthreads();

    // Persistent weight fragments (live in VGPRs across the whole T loop).
    const v16h B1_0 = make_bfrag(W1, U1, DIN, HH1, G1, 0, nl, hl);
    const v16h B1_1 = make_bfrag(W1, U1, DIN, HH1, G1, 1, nl, hl);
    const v16h B1_2 = make_bfrag(W1, U1, DIN, HH1, G1, 2, nl, hl);
    const v16h B2_0 = make_bfrag(W2, U2, HH1, HH2, G2, 0, nl, hl);
    const v16h B2_1 = make_bfrag(W2, U2, HH1, HH2, G2, 1, nl, hl);
    const v16h B3_0 = make_bfrag(W3, U3, HH2, HH3, G3, 0, nl, hl);

    // Per-lane bias scalars (C operand = bias broadcast along M).
    const float b1v0 = b1[nl];
    const float b1v1 = b1[16 + nl];
    const float b1v2 = (32 + nl) < G1 ? b1[32 + nl] : 0.0f;
    const float b2v0 = b2[nl];
    const float b2v1 = (16 + nl) < G2 ? b2[16 + nl] : 0.0f;
    const float b3v0 = b3[nl];

    // Dense head weights: lane (nl, hl) produces out[nl][2*hl + {0,1}].
    const int d0 = hl * 2, d1 = hl * 2 + 1;
    float wd0[HH3], wd1[HH3];
#pragma unroll
    for (int j = 0; j < HH3; ++j) { wd0[j] = Wd[j * HH3 + d0]; wd1[j] = Wd[j * HH3 + d1]; }
    const float bd0 = bd[d0], bd1 = bd[d1];

    // Cell states kept in registers: lane (nl, hl) owns a slice of row nl.
    float c1s[5] = {0.f, 0.f, 0.f, 0.f, 0.f};
    float c2s[4] = {0.f, 0.f, 0.f, 0.f};
    float c3s[2] = {0.f, 0.f};

    const float* xrow = x   + (size_t)(bbase + nl) * TSTEPS * DIN;
    float*       orow = out + (size_t)(bbase + nl) * TSTEPS * HH3;

    for (int t = 0; t < TSTEPS; ++t) {
        // ---- stage x_t into a1 rows (f16) ----
        if (lane < 16) {
            const float* xp = xrow + t * DIN;
#pragma unroll
            for (int d = 0; d < DIN; ++d) a1[nl * 32 + d] = (_Float16)xp[d];
            if (t + 1 < TSTEPS) __builtin_prefetch(xp + DIN);  // global_prefetch_b8
        }
        __syncthreads();

        // ---- layer 1: z1[16,40] = [x|h1] @ [W1;U1] + b1 (3 WMMA tiles) ----
        {
            v16h A = load_afrag(a1, nl, hl);
            v8f z0 = __builtin_amdgcn_wmma_f32_16x16x32_f16(false, A, false, B1_0,
                                                            (short)0, bcast8(b1v0), false, false);
            v8f z1 = __builtin_amdgcn_wmma_f32_16x16x32_f16(false, A, false, B1_1,
                                                            (short)0, bcast8(b1v1), false, false);
            v8f z2 = __builtin_amdgcn_wmma_f32_16x16x32_f16(false, A, false, B1_2,
                                                            (short)0, bcast8(b1v2), false, false);
            store_z(zbuf, z0, 0, nl, hl);
            store_z(zbuf, z1, 1, nl, hl);
            store_z(zbuf, z2, 2, nl, hl);
        }
        __syncthreads();

        // ---- layer 1 state update: lane owns j = hl*5 .. hl*5+4 of row nl ----
#pragma unroll
        for (int jj = 0; jj < 5; ++jj) {
            const int j = hl * 5 + jj;
            const float zi = zbuf[(j)            * 16 + nl];
            const float zf = zbuf[(HH1 + j)      * 16 + nl];
            const float zg = zbuf[(2 * HH1 + j)  * 16 + nl];
            const float zo = zbuf[(3 * HH1 + j)  * 16 + nl];
            const float ig = fsig(zi), fg = fsig(zf);
            const float gg = ftanh_fast(zg), og = fsig(zo);
            const float c  = fg * c1s[jj] + ig * gg;
            c1s[jj] = c;
            const _Float16 hh = (_Float16)(og * ftanh_fast(c));
            a1[nl * 32 + DIN + j] = hh;   // recurrent input for layer1 (next t)
            a2[nl * 32 + j]       = hh;   // feed-forward input for layer2
        }
        __syncthreads();

        // ---- layer 2: z2[16,28] = [h1|h2] @ [W2;U2] + b2 (2 WMMA tiles) ----
        {
            v16h A = load_afrag(a2, nl, hl);
            v8f z0 = __builtin_amdgcn_wmma_f32_16x16x32_f16(false, A, false, B2_0,
                                                            (short)0, bcast8(b2v0), false, false);
            v8f z1 = __builtin_amdgcn_wmma_f32_16x16x32_f16(false, A, false, B2_1,
                                                            (short)0, bcast8(b2v1), false, false);
            store_z(zbuf, z0, 0, nl, hl);
            store_z(zbuf, z1, 1, nl, hl);
        }
        __syncthreads();

        // ---- layer 2 state update: hl=0 -> j=0..3, hl=1 -> j=4..6 ----
#pragma unroll
        for (int jj = 0; jj < 4; ++jj) {
            const int j = hl * 4 + jj;
            if (j < HH2) {
                const float zi = zbuf[(j)            * 16 + nl];
                const float zf = zbuf[(HH2 + j)      * 16 + nl];
                const float zg = zbuf[(2 * HH2 + j)  * 16 + nl];
                const float zo = zbuf[(3 * HH2 + j)  * 16 + nl];
                const float ig = fsig(zi), fg = fsig(zf);
                const float gg = ftanh_fast(zg), og = fsig(zo);
                const float c  = fg * c2s[jj] + ig * gg;
                c2s[jj] = c;
                const _Float16 hh = (_Float16)(og * ftanh_fast(c));
                a2[nl * 32 + HH1 + j] = hh;
                a3[nl * 32 + j]       = hh;
            }
        }
        __syncthreads();

        // ---- layer 3: z3[16,16] = [h2|h3] @ [W3;U3] + b3 (1 WMMA tile) ----
        {
            v16h A = load_afrag(a3, nl, hl);
            v8f z0 = __builtin_amdgcn_wmma_f32_16x16x32_f16(false, A, false, B3_0,
                                                            (short)0, bcast8(b3v0), false, false);
            store_z(zbuf, z0, 0, nl, hl);
        }
        __syncthreads();

        // ---- layer 3 state update: hl=0 -> j=0,1; hl=1 -> j=2,3 ----
#pragma unroll
        for (int jj = 0; jj < 2; ++jj) {
            const int j = hl * 2 + jj;
            const float zi = zbuf[(j)            * 16 + nl];
            const float zf = zbuf[(HH3 + j)      * 16 + nl];
            const float zg = zbuf[(2 * HH3 + j)  * 16 + nl];
            const float zo = zbuf[(3 * HH3 + j)  * 16 + nl];
            const float ig = fsig(zi), fg = fsig(zf);
            const float gg = ftanh_fast(zg), og = fsig(zo);
            const float c  = fg * c3s[jj] + ig * gg;
            c3s[jj] = c;
            a3[nl * 32 + HH2 + j] = (_Float16)(og * ftanh_fast(c));
        }
        __syncthreads();

        // ---- dense head + output: lane (nl, hl) writes out[b=bbase+nl][t][2*hl..2*hl+1]
        {
            float h3v[HH3];
#pragma unroll
            for (int k = 0; k < HH3; ++k) h3v[k] = (float)a3[nl * 32 + HH2 + k];
            float s0 = bd0, s1 = bd1;
#pragma unroll
            for (int j = 0; j < HH3; ++j) {
                s0 = __builtin_fmaf(h3v[j], wd0[j], s0);
                s1 = __builtin_fmaf(h3v[j], wd1[j], s1);
            }
            float2 o2 = make_float2(fsig(s0), fsig(s1));
            *(float2*)(orow + t * HH3 + hl * 2) = o2;  // 8B-aligned global_store_b64
        }
        // No trailing barrier needed: next writes to a3's h3 region occur after
        // multiple syncs in the next iteration.
    }
}

extern "C" void kernel_launch(void* const* d_in, const int* in_sizes, int n_in,
                              void* d_out, int out_size, void* d_ws, size_t ws_size,
                              hipStream_t stream) {
    const float* x  = (const float*)d_in[0];
    const float* W1 = (const float*)d_in[1];
    const float* U1 = (const float*)d_in[2];
    const float* b1 = (const float*)d_in[3];
    const float* W2 = (const float*)d_in[4];
    const float* U2 = (const float*)d_in[5];
    const float* b2 = (const float*)d_in[6];
    const float* W3 = (const float*)d_in[7];
    const float* U3 = (const float*)d_in[8];
    const float* b3 = (const float*)d_in[9];
    const float* Wd = (const float*)d_in[10];
    const float* bd = (const float*)d_in[11];
    float* out = (float*)d_out;

    dim3 grid(BATCH / 16);  // 512 single-wave workgroups, 16 batch rows each
    dim3 block(32);
    lstm3_wmma_kernel<<<grid, block, 0, stream>>>(x, W1, U1, b1, W2, U2, b2,
                                                  W3, U3, b3, Wd, bd, out);
}
